// Block_47966194761867
// MI455X (gfx1250) — matrix operand order repaired and checked by reading, not compiled, not run on hardware
//
#include <hip/hip_runtime.h>

// ---------------- problem constants ----------------
#define B_   2
#define T_   2048
#define C_   768
#define H_   12
#define DH_  64
#define NI_  3072
#define M_   (B_*T_)     // 4096 rows
#define EPS_ 1e-4f
#define KSTEP_ 64        // K-slab per TDM stage

typedef __attribute__((ext_vector_type(16))) __bf16 v16bf;
typedef __attribute__((ext_vector_type(8)))  __bf16 v8bf;
typedef __attribute__((ext_vector_type(8)))  float  v8f;
typedef __attribute__((ext_vector_type(4)))  unsigned int u32x4;
typedef __attribute__((ext_vector_type(8)))  int i32x8;
typedef __attribute__((ext_vector_type(4)))  int i32x4;

// A-matrix fragment (16x32 bf16, row-major source):
// lane<16: elems 0..15 -> K=k0..k0+15
// lane>=16: K=k0+16..k0+31. Two contiguous 16B loads per lane.
static __device__ __forceinline__ v16bf load_afrag(const __bf16* rowk, int khalf) {
  v8bf lo = *(const v8bf*)(rowk + khalf*16);
  v8bf hi = *(const v8bf*)(rowk + khalf*16 + 8);
  v16bf a;
#pragma unroll
  for (int i = 0; i < 8; ++i) { a[i] = lo[i]; a[i+8] = hi[i]; }
  return a;
}

static __device__ __forceinline__ float gelu_f(float x) {
  float x3 = x*x*x;
  return 0.5f*x*(1.0f + tanhf(0.7978845608028654f*(x + 0.044715f*x3)));
}

// ---------------- Tensor Data Mover: 2-D bf16 tile -> LDS ----------------
// Packs a D# per CDNA5 ISA sec.8 and issues TENSOR_LOAD_TO_LDS.
// tile_d0 elems contiguous (dim0), tile_d1 rows, row stride = stride0 elems.
static __device__ __forceinline__ void tdm_load_2d(const void* gptr, void* lptr,
                                                   unsigned tensor_d0, unsigned tensor_d1,
                                                   unsigned tile_d0, unsigned tile_d1,
                                                   unsigned long long stride0) {
  unsigned long long ga = (unsigned long long)gptr;
  unsigned lds = (unsigned)(unsigned long long)lptr;   // low 32 bits = LDS byte offset
  u32x4 g0;
  g0[0] = 1u;                                   // count=1 (valid), no gather
  g0[1] = lds;                                  // lds_addr [63:32]
  g0[2] = (unsigned)ga;                         // global_addr [95:64]
  g0[3] = (unsigned)(ga >> 32) | (2u << 30);    // global_addr [120:96] | type=2
  i32x8 g1;
  g1[0] = (int)(1u << 16);                      // wg_mask=0, data_size=1 (2 bytes)
  g1[1] = (int)((tensor_d0 & 0xFFFFu) << 16);   // tensor_dim0[15:0] @ bits 63:48
  g1[2] = (int)((tensor_d0 >> 16) | ((tensor_d1 & 0xFFFFu) << 16)); // dim0 hi | dim1 lo
  g1[3] = (int)((tensor_d1 >> 16) | (tile_d0 << 16));               // dim1 hi | tile_dim0
  g1[4] = (int)(tile_d1 & 0xFFFFu);             // tile_dim1 (tile_dim2 = 0)
  g1[5] = (int)(unsigned)(stride0 & 0xFFFFFFFFu);          // dim0_stride [31:0]
  g1[6] = (int)(unsigned)((stride0 >> 32) & 0xFFFFu);      // dim0_stride [47:32]
  g1[7] = 0;                                    // dim1_stride unused (2-D tile)
  i32x4 gz = {0, 0, 0, 0};
#if __clang_major__ >= 23
  i32x8 gz8 = {0, 0, 0, 0, 0, 0, 0, 0};
  __builtin_amdgcn_tensor_load_to_lds(g0, g1, gz, gz, gz8, 0);
#else
  __builtin_amdgcn_tensor_load_to_lds(g0, g1, gz, gz, 0);
#endif
}

// ---------------- fp32 -> bf16 converters ----------------
__global__ __launch_bounds__(256) void k_cvt(const float* __restrict__ s,
                                             __bf16* __restrict__ d, int n) {
  int i = blockIdx.x*256 + threadIdx.x;
  if (i < n) d[i] = (__bf16)s[i];
}

// W[K][N] fp32 -> Wt[N][K] bf16 (so GEMM B tiles are K-contiguous)
__global__ __launch_bounds__(256) void k_cvt_tr(const float* __restrict__ W,
                                                __bf16* __restrict__ Wt,
                                                int K, int N) {
  int i = blockIdx.x*256 + threadIdx.x;
  if (i < K*N) {
    int k = i / N, n = i - k*N;
    Wt[(size_t)n*K + k] = (__bf16)W[i];
  }
}

// ---------------- WMMA GEMM, TDM-fed double-buffered LDS ----------------
// C[M][N] = A[M][K](bf16) * Bt[N][K](bf16)^T + bias, optional GELU.
// Block 256 = 8 waves -> 32(M) x 256(N) tile; wave -> 16x64 strip.
// Wave 0 drives the Tensor Data Mover: per 64-K stage one A-slab (32x64)
// and one B-slab (256x64) into LDS; s_wait_tensorcnt + barrier hand-off.
// qkvlay: 0 = row-major [M][N], 1 = [B,H,T,dh] head-split, 2 = [B,H,dh,T]
// (transposed-value layout so flash-attention PV B-fragments are contiguous).
__global__ __launch_bounds__(256) void k_gemm(const __bf16* __restrict__ A,
                                              const __bf16* __restrict__ Bt,
                                              const float* __restrict__ bias,
                                              float* __restrict__ Cf,
                                              __bf16* __restrict__ Cb,
                                              int Mdim, int Ndim, int Kdim,
                                              float scale, int qkvlay, int gelu) {
  __shared__ __attribute__((aligned(32))) __bf16 Abuf[2][32 * KSTEP_];
  __shared__ __attribute__((aligned(32))) __bf16 Bbuf[2][256 * KSTEP_];

  const int lane    = threadIdx.x & 31;
  const int wave    = threadIdx.x >> 5;
  const int lanecol = lane & 15;
  const int khalf   = lane >> 4;
  const int mw = wave >> 2;           // 0..1
  const int nw = wave & 3;            // 0..3
  const int mrBlock = blockIdx.y * 32;
  const int ncBlock = blockIdx.x * 256;
  const int mr = mrBlock + mw*16;
  const int nc = ncBlock + nw*64;

  const int nsteps = Kdim / KSTEP_;

  if (wave == 0) {  // prime stage 0
    tdm_load_2d(A  + (size_t)mrBlock*Kdim, &Abuf[0][0], (unsigned)Kdim, (unsigned)Mdim,
                KSTEP_, 32,  (unsigned long long)Kdim);
    tdm_load_2d(Bt + (size_t)ncBlock*Kdim, &Bbuf[0][0], (unsigned)Kdim, (unsigned)Ndim,
                KSTEP_, 256, (unsigned long long)Kdim);
  }

  v8f acc[4] = {};

  for (int s = 0; s < nsteps; ++s) {
    const int cur = s & 1;
    if (wave == 0) {
      if (s + 1 < nsteps) {           // prefetch next slab into the other buffer
        size_t koff = (size_t)(s + 1) * KSTEP_;
        tdm_load_2d(A  + (size_t)mrBlock*Kdim + koff, &Abuf[cur ^ 1][0],
                    (unsigned)Kdim, (unsigned)Mdim, KSTEP_, 32,  (unsigned long long)Kdim);
        tdm_load_2d(Bt + (size_t)ncBlock*Kdim + koff, &Bbuf[cur ^ 1][0],
                    (unsigned)Kdim, (unsigned)Ndim, KSTEP_, 256, (unsigned long long)Kdim);
        __builtin_amdgcn_s_wait_tensorcnt((short)2);  // current stage's 2 DMAs done
      } else {
        __builtin_amdgcn_s_wait_tensorcnt((short)0);
      }
    }
    __syncthreads();                  // stage `cur` ready for all waves

#pragma unroll
    for (int ks = 0; ks < KSTEP_; ks += 32) {
      v16bf a = load_afrag(&Abuf[cur][(mw*16 + lanecol)*KSTEP_ + ks], khalf);
#pragma unroll
      for (int n = 0; n < 4; ++n) {
        // B fragment: lane = column, 16 contiguous K values from LDS (32B)
        v16bf b = *(const v16bf*)(&Bbuf[cur][(nw*64 + n*16 + lanecol)*KSTEP_ + ks + khalf*16]);
        acc[n] = __builtin_amdgcn_wmma_f32_16x16x32_bf16(false, a, false, b,
                                                         (short)0, acc[n], false, false);
      }
    }
    __syncthreads();                  // everyone done reading before next overwrite
  }

  const int orow = mr + khalf*8;
#pragma unroll
  for (int n = 0; n < 4; ++n) {
    int col = nc + n*16 + lanecol;
    float bi = bias[col];
#pragma unroll
    for (int r = 0; r < 8; ++r) {
      float val = acc[n][r] + bi;
      if (gelu) val = gelu_f(val);
      int row = orow + r;
      if (Cf) Cf[(size_t)row*Ndim + col] = val;
      if (Cb) {
        size_t idx;
        if (qkvlay == 1) {        // [B,T,(H,dh)] -> [B,H,T,dh]
          int bb = row / T_, t = row - bb*T_;
          int hh = col >> 6, d = col & 63;
          idx = (((size_t)bb*H_ + hh)*T_ + t)*DH_ + d;
        } else if (qkvlay == 2) { // [B,T,(H,dh)] -> [B,H,dh,T]  (V transposed)
          int bb = row / T_, t = row - bb*T_;
          int hh = col >> 6, d = col & 63;
          idx = (((size_t)bb*H_ + hh)*DH_ + d)*T_ + t;
        } else {
          idx = (size_t)row*Ndim + col;
        }
        Cb[idx] = (__bf16)(val * scale);
      }
    }
  }
}

// ---------------- flash attention ----------------
// One wave per (b, h, 16-row q tile). 32-key tiles. q pre-scaled by 1/sqrt(dh).
// K is [B,H,T,dh] (row-major), V is pre-transposed [B,H,dh,T] so every WMMA
// B-fragment is a single contiguous 32-byte load.
__global__ __launch_bounds__(256) void k_flash(const __bf16* __restrict__ q,
                                               const __bf16* __restrict__ kmat,
                                               const __bf16* __restrict__ vt,
                                               __bf16* __restrict__ ctx) {
  __shared__ __attribute__((aligned(32))) float  Ssh[8][16*32];
  __shared__ __attribute__((aligned(32))) __bf16 Psh[8][16*32];
  __shared__ float  Alpha[8][16];
  __shared__ float  Lsh[8][16];

  const int lane    = threadIdx.x & 31;
  const int wave    = threadIdx.x >> 5;
  const int lanecol = lane & 15;
  const int khalf   = lane >> 4;

  const int gw = blockIdx.x*8 + wave;       // [0, B*H*128)
  const int qt = gw & 127;                  // T/16 = 128 q tiles
  const int bh = gw >> 7;
  const int h  = bh % H_;
  const int b  = bh / H_;
  const int qbase = qt * 16;

  const size_t headoff = ((size_t)b*H_ + h) * (size_t)T_ * DH_;
  const __bf16* qh = q    + headoff;
  const __bf16* kh = kmat + headoff;
  const __bf16* vh = vt   + headoff;        // [dh][T] within this head

  // Q fragments for d=0..31 and d=32..63 (loaded once)
  const __bf16* qrow = qh + (size_t)(qbase + lanecol) * DH_;
  v16bf aq0 = load_afrag(qrow + 0,  khalf);
  v16bf aq1 = load_afrag(qrow + 32, khalf);

  v8f o[4] = {};
  float m_st = -1e30f;  // per-row state, owned by lanes 0..15
  float l_st = 0.0f;

  for (int kb = 0; kb < qbase + 16; kb += 32) {
    // speculative prefetch of the next key/value tiles (global_prefetch_b8)
    __builtin_prefetch((const void*)(kh + (size_t)(kb + 32 + lane)*DH_), 0, 1);
    __builtin_prefetch((const void*)(vh + (size_t)lane*T_ + kb + 32), 0, 1);

    // ---- S = Q * K^T for 32 keys (two 16-col halves) ----
    v8f s0 = {}, s1 = {};
    {
      const __bf16* kcol = kh + (size_t)(kb + lanecol)*DH_ + khalf*16;
      v16bf b0 = *(const v16bf*)(kcol);
      v16bf b1 = *(const v16bf*)(kcol + 32);
      s0 = __builtin_amdgcn_wmma_f32_16x16x32_bf16(false, aq0, false, b0, (short)0, s0, false, false);
      s0 = __builtin_amdgcn_wmma_f32_16x16x32_bf16(false, aq1, false, b1, (short)0, s0, false, false);
    }
    {
      const __bf16* kcol = kh + (size_t)(kb + 16 + lanecol)*DH_ + khalf*16;
      v16bf b0 = *(const v16bf*)(kcol);
      v16bf b1 = *(const v16bf*)(kcol + 32);
      s1 = __builtin_amdgcn_wmma_f32_16x16x32_bf16(false, aq0, false, b0, (short)0, s1, false, false);
      s1 = __builtin_amdgcn_wmma_f32_16x16x32_bf16(false, aq1, false, b1, (short)0, s1, false, false);
    }

    // ---- causal mask + stage S tile to LDS ----
#pragma unroll
    for (int r = 0; r < 8; ++r) {
      int row = khalf*8 + r;
      int qi  = qbase + row;
      Ssh[wave][row*32 + lanecol]      = (kb + lanecol      <= qi) ? s0[r] : -1e30f;
      Ssh[wave][row*32 + 16 + lanecol] = (kb + 16 + lanecol <= qi) ? s1[r] : -1e30f;
    }
    asm volatile("s_wait_dscnt 0" ::: "memory");

    // ---- online softmax: one row per lane (lanes 0..15) ----
    if (lane < 16) {
      int row = lane;
      float mnew = m_st;
#pragma unroll
      for (int j = 0; j < 32; ++j) mnew = fmaxf(mnew, Ssh[wave][row*32 + j]);
      float alpha = __expf(m_st - mnew);
      float lsum = 0.0f;
#pragma unroll
      for (int j = 0; j < 32; ++j) {
        float p = __expf(Ssh[wave][row*32 + j] - mnew);
        Psh[wave][row*32 + j] = (__bf16)p;
        lsum += p;
      }
      l_st = l_st*alpha + lsum;
      m_st = mnew;
      Alpha[wave][row] = alpha;
    }
    asm volatile("s_wait_dscnt 0" ::: "memory");

    // ---- rescale running O by per-row alpha ----
#pragma unroll
    for (int r = 0; r < 8; ++r) {
      float al = Alpha[wave][khalf*8 + r];
      o[0][r] *= al; o[1][r] *= al; o[2][r] *= al; o[3][r] *= al;
    }

    // ---- P fragment (A layout, 16x32): LDS layout matches A-fragment
    //      order, so this is 2x ds_load_b128 instead of 16 scalar loads ----
    v16bf pa = load_afrag(&Psh[wave][lanecol*32], khalf);

    // ---- O += P * V (V tile 32x64 as 4 B-fragments; V^T is key-contiguous) ----
#pragma unroll
    for (int n = 0; n < 4; ++n) {
      int d = n*16 + lanecol;
      v16bf bv = *(const v16bf*)(vh + (size_t)d*T_ + kb + khalf*16);
      o[n] = __builtin_amdgcn_wmma_f32_16x16x32_bf16(false, pa, false, bv, (short)0, o[n], false, false);
    }
  }

  // ---- finalize: divide by row sums, write ctx [B,T,C] bf16 ----
  if (lane < 16) Lsh[wave][lane] = l_st;
  asm volatile("s_wait_dscnt 0" ::: "memory");
#pragma unroll
  for (int r = 0; r < 8; ++r) {
    int row = khalf*8 + r;
    float inv = 1.0f / Lsh[wave][row];
    int t = qbase + row;
#pragma unroll
    for (int n = 0; n < 4; ++n) {
      int d = n*16 + lanecol;
      ctx[((size_t)b*T_ + t)*C_ + h*DH_ + d] = (__bf16)(o[n][r] * inv);
    }
  }
}

// ---------------- residual + LayerNorm (ddof=1) ----------------
// One block (256 threads) per row of 768.
__global__ __launch_bounds__(256) void k_add_ln(const float* __restrict__ x,
                                                const float* __restrict__ y,
                                                const float* __restrict__ sc,
                                                const float* __restrict__ bi,
                                                float* __restrict__ outf,
                                                __bf16* __restrict__ outb) {
  __shared__ float red[256];
  const int row = blockIdx.x;
  const int tid = threadIdx.x;
  const float* xr = x + (size_t)row*C_;
  const float* yr = y + (size_t)row*C_;

  float v[3];
  float sum = 0.f;
#pragma unroll
  for (int i = 0; i < 3; ++i) {
    int c = tid + i*256;
    v[i] = xr[c] + yr[c];
    sum += v[i];
  }
  red[tid] = sum; __syncthreads();
  for (int s = 128; s > 0; s >>= 1) { if (tid < s) red[tid] += red[tid + s]; __syncthreads(); }
  float mean = red[0] / (float)C_;
  __syncthreads();

  float sq = 0.f;
#pragma unroll
  for (int i = 0; i < 3; ++i) { float d = v[i] - mean; sq += d*d; }
  red[tid] = sq; __syncthreads();
  for (int s = 128; s > 0; s >>= 1) { if (tid < s) red[tid] += red[tid + s]; __syncthreads(); }
  float var  = red[0] / (float)(C_ - 1);   // unbiased, matches torch.var default
  float rstd = rsqrtf(var + EPS_);

#pragma unroll
  for (int i = 0; i < 3; ++i) {
    int c = tid + i*256;
    float ov = sc[c]*((v[i] - mean)*rstd) + bi[c];
    outf[(size_t)row*C_ + c] = ov;
    if (outb) outb[(size_t)row*C_ + c] = (__bf16)ov;
  }
}

// ---------------- host orchestration ----------------
extern "C" void kernel_launch(void* const* d_in, const int* in_sizes, int n_in,
                              void* d_out, int out_size, void* d_ws, size_t ws_size,
                              hipStream_t stream) {
  (void)in_sizes; (void)n_in; (void)out_size; (void)ws_size;
  const float* x   = (const float*)d_in[0];
  const float* Wq  = (const float*)d_in[1];
  const float* bq  = (const float*)d_in[2];
  const float* Wk  = (const float*)d_in[3];
  const float* bk  = (const float*)d_in[4];
  const float* Wv  = (const float*)d_in[5];
  const float* bv  = (const float*)d_in[6];
  const float* Wo  = (const float*)d_in[7];
  const float* bo  = (const float*)d_in[8];
  const float* l1s = (const float*)d_in[9];
  const float* l1b = (const float*)d_in[10];
  const float* W1  = (const float*)d_in[11];
  const float* b1  = (const float*)d_in[12];
  const float* W2  = (const float*)d_in[13];
  const float* b2  = (const float*)d_in[14];
  const float* l2s = (const float*)d_in[15];
  const float* l2b = (const float*)d_in[16];
  float* out = (float*)d_out;

  char* ws = (char*)d_ws;
  size_t off = 0;
  auto alloc = [&](size_t bytes) -> char* {
    char* p = ws + off;
    off += (bytes + 255) & ~(size_t)255;
    return p;
  };

  __bf16* XB   = (__bf16*)alloc((size_t)M_*C_*2);
  __bf16* WQT  = (__bf16*)alloc((size_t)C_*C_*2);
  __bf16* WKT  = (__bf16*)alloc((size_t)C_*C_*2);
  __bf16* WVT  = (__bf16*)alloc((size_t)C_*C_*2);
  __bf16* WOT  = (__bf16*)alloc((size_t)C_*C_*2);
  __bf16* W1T  = (__bf16*)alloc((size_t)NI_*C_*2);
  __bf16* W2T  = (__bf16*)alloc((size_t)C_*NI_*2);
  __bf16* QB   = (__bf16*)alloc((size_t)M_*C_*2);
  __bf16* KB   = (__bf16*)alloc((size_t)M_*C_*2);
  __bf16* VTB  = (__bf16*)alloc((size_t)M_*C_*2);   // V transposed [B,H,dh,T]
  __bf16* CTXB = (__bf16*)alloc((size_t)M_*C_*2);
  __bf16* HB   = (__bf16*)alloc((size_t)M_*C_*2);
  __bf16* ACT  = (__bf16*)alloc((size_t)M_*NI_*2);
  float*  AOUT = (float*)alloc((size_t)M_*C_*4);
  float*  HF   = (float*)alloc((size_t)M_*C_*4);
  float*  MOUT = (float*)alloc((size_t)M_*C_*4);

  // 1) bf16 conversions (weights transposed to [N][K])
  k_cvt<<<dim3((M_*C_ + 255)/256), dim3(256), 0, stream>>>(x, XB, M_*C_);
  k_cvt_tr<<<dim3((C_*C_ + 255)/256),  dim3(256), 0, stream>>>(Wq, WQT, C_, C_);
  k_cvt_tr<<<dim3((C_*C_ + 255)/256),  dim3(256), 0, stream>>>(Wk, WKT, C_, C_);
  k_cvt_tr<<<dim3((C_*C_ + 255)/256),  dim3(256), 0, stream>>>(Wv, WVT, C_, C_);
  k_cvt_tr<<<dim3((C_*C_ + 255)/256),  dim3(256), 0, stream>>>(Wo, WOT, C_, C_);
  k_cvt_tr<<<dim3((C_*NI_ + 255)/256), dim3(256), 0, stream>>>(W1, W1T, C_, NI_);
  k_cvt_tr<<<dim3((NI_*C_ + 255)/256), dim3(256), 0, stream>>>(W2, W2T, NI_, C_);

  // 2) QKV projections (q scaled by 1/sqrt(dh)=0.125; V stored transposed)
  dim3 gqkv(C_/256, M_/32);
  k_gemm<<<gqkv, dim3(256), 0, stream>>>(XB, WQT, bq, nullptr, QB,  M_, C_, C_, 0.125f, 1, 0);
  k_gemm<<<gqkv, dim3(256), 0, stream>>>(XB, WKT, bk, nullptr, KB,  M_, C_, C_, 1.0f,   1, 0);
  k_gemm<<<gqkv, dim3(256), 0, stream>>>(XB, WVT, bv, nullptr, VTB, M_, C_, C_, 1.0f,   2, 0);

  // 3) flash attention: B*H*(T/16) = 3072 waves -> 384 blocks of 8 waves
  k_flash<<<dim3((B_*H_*(T_/16))/8), dim3(256), 0, stream>>>(QB, KB, VTB, CTXB);

  // 4) output projection (fp32 out)
  k_gemm<<<gqkv, dim3(256), 0, stream>>>(CTXB, WOT, bo, AOUT, nullptr, M_, C_, C_, 1.0f, 0, 0);

  // 5) h = LN1(x + attn_out)  (fp32 + bf16 copies)
  k_add_ln<<<dim3(M_), dim3(256), 0, stream>>>(x, AOUT, l1s, l1b, HF, HB);

  // 6) MLP: gelu(h@W1+b1) @ W2 + b2
  dim3 gm1(NI_/256, M_/32);
  k_gemm<<<gm1,  dim3(256), 0, stream>>>(HB, W1T, b1, nullptr, ACT, M_, NI_, C_, 1.0f, 0, 1);
  k_gemm<<<gqkv, dim3(256), 0, stream>>>(ACT, W2T, b2, MOUT, nullptr, M_, C_, NI_, 1.0f, 0, 0);

  // 7) out = LN2(h + mlp_out)
  k_add_ln<<<dim3(M_), dim3(256), 0, stream>>>(HF, MOUT, l2s, l2b, out, nullptr);
}
